// AutoNER_91173565759685
// MI455X (gfx1250) — matrix operand order repaired and checked by reading, not compiled, not run on hardware
//
#include <hip/hip_runtime.h>
#include <math.h>

typedef __attribute__((ext_vector_type(16))) _Float16 v16h;
typedef __attribute__((ext_vector_type(8)))  float    v8f;
typedef __attribute__((ext_vector_type(4)))  int      v4i;

// ---------------- problem constants ----------------
#define B_   64
#define S_   200
#define TC_  16
#define NT_  (B_*S_)          // 12800 words
#define DW_  200
#define DC_  20
#define CH_  50               // char hidden
#define WH_  100              // word hidden
#define HID_ 256

// padded dims (K padded to mult of 32, N to mult of 16)
#define KC0_  32              // char l0 input (20 -> 32)
#define KC1_  128             // char l1 input (100 -> 128)
#define CHP_  64              // char H pad (50 -> 64)
#define CNG_  208             // char 4H pad (200 -> 208)
#define KW0_  416             // word l0 input (400 -> 416)
#define KW1_  224             // word l1+ input (200 -> 224)
#define WHP_  128             // word H pad (100 -> 128)
#define WNG_  400             // word 4H (400, already mult 16)

// ---------------- async global->LDS copy (gfx1250 ASYNCcnt path) -----------
#if defined(__gfx1250__) && __has_builtin(__builtin_amdgcn_global_load_async_to_lds_b128)
#define HAS_ASYNC_LDS 1
#else
#define HAS_ASYNC_LDS 0
#endif

#if HAS_ASYNC_LDS
typedef __attribute__((address_space(1))) v4i gl_v4i;   // global 128-bit chunk
typedef __attribute__((address_space(3))) v4i lds_v4i;  // LDS 128-bit chunk
#endif

__device__ __forceinline__ void copy16_to_lds(const _Float16* gsrc, _Float16* ldst) {
#if HAS_ASYNC_LDS
  __builtin_amdgcn_global_load_async_to_lds_b128(
      (gl_v4i*)(size_t)gsrc,
      (lds_v4i*)(unsigned)(size_t)ldst, 0, 0);
#else
  *reinterpret_cast<float4*>(ldst) = *reinterpret_cast<const float4*>(gsrc);
#endif
}
__device__ __forceinline__ void wait_async_lds() {
#if HAS_ASYNC_LDS
  asm volatile("s_wait_asynccnt 0" ::: "memory");
#endif
}

// ---------------- WMMA 16x16 tile, one K-block of 32 (f16 in, f32 acc) ----
// A: row-major tile, element (m,k) at A[m*lda + k]
// Bt: row-major weight [N,K]; logical B[k][n] = Bt[n*ldb + k]
__device__ __forceinline__ void wmma_kblock(v8f& acc,
                                            const _Float16* A, int lda,
                                            const _Float16* Bt, int ldb) {
  int lane = threadIdx.x & 31;
  int m  = lane & 15;
  int ks = (lane < 16) ? 0 : 8;    // A: K groups {ks..ks+7, ks+16..ks+23}
  int kb = (lane < 16) ? 0 : 8;    // B: K groups {kb..kb+7, kb+16..kb+23}
  union { v16h v; float4 q[2]; } a, b;
  const _Float16* ap = A + m * lda + ks;
  a.q[0] = *reinterpret_cast<const float4*>(ap);
  a.q[1] = *reinterpret_cast<const float4*>(ap + 16);
  const _Float16* bp = Bt + m * ldb + kb;
  b.q[0] = *reinterpret_cast<const float4*>(bp);
  b.q[1] = *reinterpret_cast<const float4*>(bp + 16);
  acc = __builtin_amdgcn_wmma_f32_16x16x32_f16(false, a.v, false, b.v,
                                               (short)0, acc, false, false);
}

// ---------------- small utility kernels ----------------
__global__ void zero16_k(_Float16* p, int n) {
  int i = blockIdx.x * blockDim.x + threadIdx.x;
  if (i < n) p[i] = (_Float16)0.f;
}
__global__ void zero32_k(float* p, int n) {
  int i = blockIdx.x * blockDim.x + threadIdx.x;
  if (i < n) p[i] = 0.f;
}
// f32 [srows,scols] -> zero-padded f16 [drows,dcols]
__global__ void pack_f16_k(_Float16* dst, const float* src,
                           int srows, int scols, int drows, int dcols) {
  int i = blockIdx.x * blockDim.x + threadIdx.x;
  if (i >= drows * dcols) return;
  int r = i / dcols, c = i % dcols;
  float v = (r < srows && c < scols) ? src[r * scols + c] : 0.f;
  dst[i] = (_Float16)v;
}
__global__ void pack_bias_k(float* dst, const float* b0, const float* b1,
                            int ng, int ngpad) {
  int i = blockIdx.x * blockDim.x + threadIdx.x;
  if (i >= ngpad) return;
  dst[i] = (i < ng) ? (b0[i] + b1[i]) : 0.f;
}
// gather char embeddings -> f16 [NT*TC, 32]
__global__ void gather_char_k(_Float16* x0, const float* char_emb,
                              const int* cids, int total_rows) {
  int idx = blockIdx.x * blockDim.x + threadIdx.x;
  if (idx >= total_rows * KC0_) return;
  int col = idx & (KC0_ - 1);
  int r   = idx >> 5;
  float v = 0.f;
  if (col < DC_) {
    int id = cids[r] & 255;
    v = char_emb[id * DC_ + col];
  }
  x0[idx] = (_Float16)v;
}
// word input: [word_emb(200) | char first(100) | char last(100) | pad16]
__global__ void build_word_input_k(_Float16* xw, const float* word_emb,
                                   const int* ids, const _Float16* c2,
                                   const int* clen) {
  int i = blockIdx.x;  // word row 0..NT-1
  int id = ids[i]; if (id < 0) id = 0;
  int L = clen[i]; int tl = L - 1; if (tl < 0) tl = 0; if (tl > TC_ - 1) tl = TC_ - 1;
  for (int col = threadIdx.x; col < KW0_; col += blockDim.x) {
    float v = 0.f;
    if (col < DW_)            v = word_emb[(size_t)id * DW_ + col];
    else if (col < DW_ + 100) v = (float)c2[((size_t)i * TC_ + 0)  * KC1_ + (col - DW_)];
    else if (col < DW_ + 200) v = (float)c2[((size_t)i * TC_ + tl) * KC1_ + (col - DW_ - 100)];
    xw[(size_t)i * KW0_ + col] = (_Float16)v;
  }
}
// zero rows of final word output where s >= slength[b]
__global__ void mask_rows_k(_Float16* Y, const int* slength, int total) {
  int i = blockIdx.x * blockDim.x + threadIdx.x;
  if (i >= total) return;
  int row = i / KW1_;
  int b = row / S_, s = row % S_;
  if (s >= slength[b]) Y[i] = (_Float16)0.f;
}

// ---------------- fused persistent LSTM direction kernel --------------------
// One launch = one (layer, direction). X: f16 [n, T, ldx]; Y gets h at col ycol0.
// Per step: X_t staged to LDS (async DMA), gates = X_t@Wih^T + h@Whh^T (WMMA),
// then elementwise cell update with mask t < lens[row].
__global__ void lstm_dir_k(const _Float16* __restrict__ X, int ldx, int Kx,
                           const _Float16* __restrict__ Wih,   // [Ngpad, ldx]
                           const _Float16* __restrict__ Whh,   // [Ngpad, Hpad]
                           const float*    __restrict__ bias,  // [Ngpad]
                           const int*      __restrict__ lens,
                           int T, int H, int Hpad, int Ngpad,
                           _Float16* __restrict__ Y, int ldy, int ycol0, int rev) {
  extern __shared__ char smem[];
  const int ROWS = 32;
  float*    c_lds = (float*)smem;                                           // ROWS*Hpad f32
  _Float16* h_lds = (_Float16*)(smem + (size_t)ROWS * Hpad * 4);            // ROWS*Hpad f16
  _Float16* g_lds = (_Float16*)(smem + (size_t)ROWS * Hpad * 6);            // ROWS*Ngpad f16
  _Float16* x_lds = (_Float16*)(smem + (size_t)ROWS * (Hpad * 6 + Ngpad * 2)); // ROWS*ldx f16

  int row0 = blockIdx.x * ROWS;
  int wid = threadIdx.x >> 5, lane = threadIdx.x & 31;
  int nwaves = blockDim.x >> 5;
  for (int i = threadIdx.x; i < ROWS * Hpad; i += blockDim.x) {
    c_lds[i] = 0.f; h_lds[i] = (_Float16)0.f;
  }
  __syncthreads();

  const int Mt = ROWS / 16, Nt = Ngpad / 16;
  const int cpr = (ldx * 2) >> 4;  // 16B chunks per activation row
  for (int step = 0; step < T; ++step) {
    int t = rev ? (T - 1 - step) : step;
    // ---- phase 0: stage X_t (ROWS x ldx) into LDS via async global->LDS ----
    for (int i = threadIdx.x; i < ROWS * cpr; i += blockDim.x) {
      int r = i / cpr, c = i % cpr;
      const _Float16* gsrc = X + ((size_t)(row0 + r) * T + t) * ldx + c * 8;
      copy16_to_lds(gsrc, x_lds + r * ldx + c * 8);
    }
    // prefetch next timestep's activations into cache while DMA completes
    if (step + 1 < T) {
      int t2 = rev ? (T - 2 - step) : (step + 1);
      const char* pf = (const char*)(X + ((size_t)(row0 + (threadIdx.x & 31)) * T + t2) * ldx);
      __builtin_prefetch(pf, 0, 1);
    }
    wait_async_lds();
    __syncthreads();
    // ---- phase 1: gate GEMM (WMMA), wave-uniform tile loop ----
    for (int tile = wid; tile < Mt * Nt; tile += nwaves) {
      int mt = tile / Nt, nt = tile % Nt;
      v8f acc = {};
      const _Float16* Abase = x_lds + mt * 16 * ldx;
      const _Float16* Bbase = Wih + (size_t)(nt * 16) * ldx;
      for (int kb = 0; kb < Kx; kb += 32)
        wmma_kblock(acc, Abase + kb, ldx, Bbase + kb, ldx);
      const _Float16* Hbase = h_lds + mt * 16 * Hpad;
      const _Float16* B2    = Whh + (size_t)(nt * 16) * Hpad;
      for (int kb = 0; kb < Hpad; kb += 32)
        wmma_kblock(acc, Hbase + kb, Hpad, B2 + kb, Hpad);
      int nn = lane & 15, mb = (lane < 16) ? 0 : 8;
      #pragma unroll
      for (int v = 0; v < 8; ++v)
        g_lds[(mt * 16 + mb + v) * Ngpad + nt * 16 + nn] = (_Float16)acc[v];
    }
    __syncthreads();
    // ---- phase 2: elementwise LSTM cell update ----
    for (int e = threadIdx.x; e < ROWS * H; e += blockDim.x) {
      int r = e / H, j = e % H;
      int grow = row0 + r;
      float gi = (float)g_lds[r * Ngpad + j]         + bias[j];
      float gf = (float)g_lds[r * Ngpad + H + j]     + bias[H + j];
      float gg = (float)g_lds[r * Ngpad + 2 * H + j] + bias[2 * H + j];
      float go = (float)g_lds[r * Ngpad + 3 * H + j] + bias[3 * H + j];
      float si = 1.f / (1.f + expf(-gi));
      float sf = 1.f / (1.f + expf(-gf));
      float so = 1.f / (1.f + expf(-go));
      float tg = tanhf(gg);
      float cold = c_lds[r * Hpad + j];
      float c2v = sf * cold + si * tg;
      float h2v = so * tanhf(c2v);
      float hold = (float)h_lds[r * Hpad + j];
      bool msk = t < lens[grow];
      float cn = msk ? c2v : cold;
      float hn = msk ? h2v : hold;
      c_lds[r * Hpad + j] = cn;
      h_lds[r * Hpad + j] = (_Float16)hn;
      Y[((size_t)grow * T + t) * ldy + ycol0 + j] = (_Float16)hn;
    }
    __syncthreads();
  }
}

// ---------------- WMMA GEMM with fused bias + relu (span hidden layer) ------
// One block per 16-row M-tile: A strip staged to LDS once, 8 waves cover the
// 16 N-tiles from LDS.  K = KW1_ (224), N = HID_ (256).
__global__ void wmma_gemm_bias_relu_k(const _Float16* __restrict__ A, int lda,
                                      const _Float16* __restrict__ Bt, int ldb,
                                      const float* __restrict__ bias,
                                      _Float16* __restrict__ C, int ldc,
                                      int Ntiles, int K) {
  __shared__ _Float16 a_lds[16 * KW1_];
  int mt = blockIdx.x;
  int wid = threadIdx.x >> 5, lane = threadIdx.x & 31;
  int nwaves = blockDim.x >> 5;
  int cpr = (K * 2) >> 4;
  for (int i = threadIdx.x; i < 16 * cpr; i += blockDim.x) {
    int r = i / cpr, c = i % cpr;
    copy16_to_lds(A + (size_t)(mt * 16 + r) * lda + c * 8, a_lds + r * K + c * 8);
  }
  wait_async_lds();
  __syncthreads();
  for (int nt = wid; nt < Ntiles; nt += nwaves) {
    v8f acc = {};
    for (int kb = 0; kb < K; kb += 32)
      wmma_kblock(acc, a_lds + kb, K, Bt + (size_t)(nt * 16) * ldb + kb, ldb);
    int nn = lane & 15, mb = (lane < 16) ? 0 : 8;
    #pragma unroll
    for (int v = 0; v < 8; ++v) {
      float val = acc[v] + bias[nt * 16 + nn];
      val = val > 0.f ? val : 0.f;
      C[(size_t)(mt * 16 + mb + v) * ldc + nt * 16 + nn] = (_Float16)val;
    }
  }
}

// ---------------- 2-class logits + log-softmax + NLL accumulation -----------
__global__ void span_loss_k(const _Float16* __restrict__ h1,
                            const float* __restrict__ W2, const float* __restrict__ b2,
                            const int* __restrict__ label, float* scal, int rows) {
  int i = blockIdx.x * blockDim.x + threadIdx.x;
  if (i >= rows) return;
  const _Float16* hr = h1 + (size_t)i * HID_;
  float z0 = b2[0], z1 = b2[1];
  for (int k = 0; k < HID_; ++k) {
    float hv = (float)hr[k];
    z0 += hv * W2[k];
    z1 += hv * W2[HID_ + k];
  }
  float mx = fmaxf(z0, z1);
  float lse = mx + logf(expf(z0 - mx) + expf(z1 - mx));
  int lb = label[i];
  bool valid = (lb != -100);
  int l = valid ? lb : 0;
  float nll = lse - (l == 0 ? z0 : z1);
  if (valid) {
    atomicAdd(scal + 0, nll);
    atomicAdd(scal + 1, 1.f);
  }
}
__global__ void finalize_k(const float* scal, float* out) {
  out[0] = scal[0] / fmaxf(scal[1], 1.f);
  out[1] = 0.f;
}

// ---------------- host: orchestration ----------------
static inline dim3 g1(long n, int b) { return dim3((unsigned)((n + b - 1) / b)); }

extern "C" void kernel_launch(void* const* d_in, const int* in_sizes, int n_in,
                              void* d_out, int out_size, void* d_ws, size_t ws_size,
                              hipStream_t stream) {
  (void)in_sizes; (void)n_in; (void)out_size; (void)ws_size;
  const float* word_emb = (const float*)d_in[0];
  const float* char_emb = (const float*)d_in[1];
  // char_lstm: layers at d_in[2 + 8*l], 8 tensors each (Wif,Whf,bif,bhf,Wib,Whb,bib,bhb)
  // word_lstm: layers at d_in[18 + 8*l]
  const float* span_W1 = (const float*)d_in[50];
  const float* span_b1 = (const float*)d_in[51];
  const float* span_W2 = (const float*)d_in[52];
  const float* span_b2 = (const float*)d_in[53];
  const int* input_ids          = (const int*)d_in[54];
  const int* input_char_ids     = (const int*)d_in[55];
  const int* input_char_lengths = (const int*)d_in[56];
  const int* slength            = (const int*)d_in[57];
  const int* span_label         = (const int*)d_in[58];

  size_t off = 0;
  auto alloc = [&](size_t bytes) -> void* {
    void* p = (char*)d_ws + off;
    off += (bytes + 255) & ~(size_t)255;
    return p;
  };
  _Float16* x0c = (_Float16*)alloc((size_t)NT_ * TC_ * KC0_ * 2);
  _Float16* c1  = (_Float16*)alloc((size_t)NT_ * TC_ * KC1_ * 2);
  _Float16* c2  = (_Float16*)alloc((size_t)NT_ * TC_ * KC1_ * 2);
  _Float16* xw  = (_Float16*)alloc((size_t)NT_ * KW0_ * 2);
  _Float16* yw0 = (_Float16*)alloc((size_t)NT_ * KW1_ * 2);
  _Float16* yw1 = (_Float16*)alloc((size_t)NT_ * KW1_ * 2);
  _Float16* h1  = (_Float16*)alloc((size_t)NT_ * HID_ * 2);
  _Float16* W1h = (_Float16*)alloc((size_t)HID_ * KW1_ * 2);
  float*    scal = (float*)alloc(2 * 4);
  _Float16 *cWih[2][2], *cWhh[2][2]; float* cB[2][2];
  for (int l = 0; l < 2; ++l)
    for (int d = 0; d < 2; ++d) {
      cWih[l][d] = (_Float16*)alloc((size_t)CNG_ * (l ? KC1_ : KC0_) * 2);
      cWhh[l][d] = (_Float16*)alloc((size_t)CNG_ * CHP_ * 2);
      cB[l][d]   = (float*)alloc((size_t)CNG_ * 4);
    }
  _Float16 *wWih[4][2], *wWhh[4][2]; float* wB[4][2];
  for (int l = 0; l < 4; ++l)
    for (int d = 0; d < 2; ++d) {
      wWih[l][d] = (_Float16*)alloc((size_t)WNG_ * (l ? KW1_ : KW0_) * 2);
      wWhh[l][d] = (_Float16*)alloc((size_t)WNG_ * WHP_ * 2);
      wB[l][d]   = (float*)alloc((size_t)WNG_ * 4);
    }

  // ---- init / pack ----
  zero32_k<<<1, 2, 0, stream>>>(scal, 2);
  zero16_k<<<g1((long)NT_ * TC_ * KC1_, 256), 256, 0, stream>>>(c1, NT_ * TC_ * KC1_);
  zero16_k<<<g1((long)NT_ * TC_ * KC1_, 256), 256, 0, stream>>>(c2, NT_ * TC_ * KC1_);
  zero16_k<<<g1((long)NT_ * KW1_, 256), 256, 0, stream>>>(yw0, NT_ * KW1_);
  zero16_k<<<g1((long)NT_ * KW1_, 256), 256, 0, stream>>>(yw1, NT_ * KW1_);

  for (int l = 0; l < 2; ++l)
    for (int d = 0; d < 2; ++d) {
      int base = 2 + 8 * l + 4 * d;
      int kin = l ? 100 : DC_, kpad = l ? KC1_ : KC0_;
      pack_f16_k<<<g1((long)CNG_ * kpad, 256), 256, 0, stream>>>(
          cWih[l][d], (const float*)d_in[base + 0], 4 * CH_, kin, CNG_, kpad);
      pack_f16_k<<<g1((long)CNG_ * CHP_, 256), 256, 0, stream>>>(
          cWhh[l][d], (const float*)d_in[base + 1], 4 * CH_, CH_, CNG_, CHP_);
      pack_bias_k<<<1, 256, 0, stream>>>(
          cB[l][d], (const float*)d_in[base + 2], (const float*)d_in[base + 3], 4 * CH_, CNG_);
    }
  for (int l = 0; l < 4; ++l)
    for (int d = 0; d < 2; ++d) {
      int base = 18 + 8 * l + 4 * d;
      int kin = l ? 200 : 400, kpad = l ? KW1_ : KW0_;
      pack_f16_k<<<g1((long)WNG_ * kpad, 256), 256, 0, stream>>>(
          wWih[l][d], (const float*)d_in[base + 0], 4 * WH_, kin, WNG_, kpad);
      pack_f16_k<<<g1((long)WNG_ * WHP_, 256), 256, 0, stream>>>(
          wWhh[l][d], (const float*)d_in[base + 1], 4 * WH_, WH_, WNG_, WHP_);
      pack_bias_k<<<2, 256, 0, stream>>>(
          wB[l][d], (const float*)d_in[base + 2], (const float*)d_in[base + 3], 4 * WH_, WNG_);
    }
  pack_f16_k<<<g1((long)HID_ * KW1_, 256), 256, 0, stream>>>(
      W1h, span_W1, HID_, 2 * WH_, HID_, KW1_);

  // ---- char BiLSTM ----
  gather_char_k<<<g1((long)NT_ * TC_ * KC0_, 256), 256, 0, stream>>>(
      x0c, char_emb, input_char_ids, NT_ * TC_);
  for (int l = 0; l < 2; ++l)
    for (int d = 0; d < 2; ++d) {
      int ldx = l ? KC1_ : KC0_;
      size_t smemC = 32 * (CHP_ * 6 + CNG_ * 2 + ldx * 2);
      lstm_dir_k<<<NT_ / 32, 256, smemC, stream>>>(
          l ? c1 : x0c, ldx, ldx,
          cWih[l][d], cWhh[l][d], cB[l][d], input_char_lengths,
          TC_, CH_, CHP_, CNG_, l ? c2 : c1, KC1_, d * CH_, d);
    }

  // ---- word input features ----
  build_word_input_k<<<NT_, 256, 0, stream>>>(xw, word_emb, input_ids, c2,
                                              input_char_lengths);

  // ---- word BiLSTM (ping-pong yw0/yw1) ----
  for (int l = 0; l < 4; ++l) {
    const _Float16* Xl = (l == 0) ? xw : ((l % 2) ? yw0 : yw1);
    _Float16* Yl = (l % 2 == 0) ? yw0 : yw1;
    int ldx = l ? KW1_ : KW0_;
    size_t smemW = 32 * (WHP_ * 6 + WNG_ * 2 + ldx * 2);
    for (int d = 0; d < 2; ++d)
      lstm_dir_k<<<B_ / 32, 256, smemW, stream>>>(
          Xl, ldx, ldx,
          wWih[l][d], wWhh[l][d], wB[l][d], slength,
          S_, WH_, WHP_, WNG_, Yl, KW1_, d * WH_, d);
  }

  // ---- mask + span head ----
  mask_rows_k<<<g1((long)NT_ * KW1_, 256), 256, 0, stream>>>(yw1, slength,
                                                             NT_ * KW1_);
  wmma_gemm_bias_relu_k<<<NT_ / 16, 256, 0, stream>>>(
      yw1, KW1_, W1h, KW1_, span_b1, h1, HID_, HID_ / 16, KW1_);
  span_loss_k<<<g1(NT_, 256), 256, 0, stream>>>(h1, span_W2, span_b2, span_label,
                                                scal, NT_);
  finalize_k<<<1, 1, 0, stream>>>(scal, (float*)d_out);
}
